// DeepTypedGraphNet_73615739454080
// MI455X (gfx1250) — matrix operand order repaired and checked by reading, not compiled, not run on hardware
//
#include <hip/hip_runtime.h>
#include <hip/hip_bf16.h>

#define DEV __device__ __forceinline__

typedef __bf16 bf16_t;
typedef __attribute__((ext_vector_type(16))) __bf16 v16bf;
typedef __attribute__((ext_vector_type(8)))  __bf16 v8bf;
typedef __attribute__((ext_vector_type(8)))  float  v8f;

static constexpr int NNODE = 100000;
static constexpr int NEDGE = 300000;
static constexpr int LAT   = 256;
static constexpr int STEPS = 6;

struct AStage { v8bf v[4]; };   // 32 bf16 staged in regs
struct BStage { v8bf v[8]; };   // 64 bf16 staged in regs

// ---------------- WMMA fragment helpers (LDS row stride = 64 bf16) ----------------
// ISA 16-bit A 16x32 layout: lanes 0-15 -> M=lane, K = {0..7, 16..23};
//                            lanes16-31 -> M=lane-16, K = {8..15, 24..31}.
DEV v16bf frag_a64(const bf16_t* Alds, int mbase, int koff, int lane) {
  int m  = mbase + (lane & 15);
  int kb = koff + (lane >> 4) * 8;        // koff + {0,8}
  const bf16_t* p = Alds + m * 64 + kb;   // 16B aligned
  v8bf lo = *(const v8bf*)(p);            // K = kb .. kb+7
  v8bf hi = *(const v8bf*)(p + 16);       // K = kb+16 .. kb+23
  v16bf a;
#pragma unroll
  for (int i = 0; i < 8; ++i) { a[i] = lo[i]; a[i + 8] = hi[i]; }
  return a;
}

// B 32x16 layout (matches A): lanes 0-15 -> N=lane, K = {0..7, 16..23}; lanes16-31 -> K = {8..15, 24..31}.
DEV v16bf frag_b64(const bf16_t* Blds, int t, int koff, int lane) {
  int n  = t * 16 + (lane & 15);
  int kh = koff + (lane >> 4) * 8;        // koff + {0,8}
  const bf16_t* p = Blds + n * 64 + kh;   // 16B aligned
  v8bf lo = *(const v8bf*)(p);
  v8bf hi = *(const v8bf*)(p + 16);
  v16bf b;
#pragma unroll
  for (int i = 0; i < 8; ++i) { b[i] = lo[i]; b[i + 8] = hi[i]; }
  return b;
}

DEV v8f wmma_bf16(v16bf a, v16bf b, v8f c) {
  return __builtin_amdgcn_wmma_f32_16x16x32_bf16(false, a, false, b, (short)0, c,
                                                 false, false);
}

// Distance-2 rotating prefetch: b0/b1 live across each WMMA, so their
// ds_load_b128 pairs stay in flight behind the matrix pipe. Fully unrolled,
// the rotation is SSA renaming (no v_mov), and regalloc cannot coalesce the
// fragments because the live ranges overlap a WMMA.
template <int NT>
DEV void mma_row(v16bf a, const bf16_t* Blds, int koff, int lane, v8f* acc) {
  v16bf b0 = frag_b64(Blds, 0, koff, lane);
  v16bf b1 = frag_b64(Blds, 1, koff, lane);
#pragma unroll
  for (int t = 0; t < NT; ++t) {
    v16bf bn = b1;
    if (t + 2 < NT) bn = frag_b64(Blds, t + 2, koff, lane);
    acc[t] = wmma_bf16(a, b0, acc[t]);
    b0 = b1;
    b1 = bn;
  }
}

DEV void stage_store4(bf16_t* dst, const AStage& s) {
#pragma unroll
  for (int j = 0; j < 4; ++j) *(v8bf*)(dst + j * 8) = s.v[j];
}
DEV void stage_store8(bf16_t* dst, const BStage& s) {
#pragma unroll
  for (int j = 0; j < 8; ++j) *(v8bf*)(dst + j * 8) = s.v[j];
}

// ---------------- weight convert: fp32 W[K][N] -> bf16 Wt[N][K] ----------------
__global__ void k_w2bf16t(const float* __restrict__ W, bf16_t* __restrict__ Wt,
                          int K, int N) {
  int idx = blockIdx.x * 256 + threadIdx.x;
  if (idx >= K * N) return;
  int k = idx / N, n = idx - k * N;
  Wt[(size_t)n * K + k] = (bf16_t)W[idx];
}

__global__ void k_fill0(float* __restrict__ p, int n) {
  int i = blockIdx.x * 256 + threadIdx.x;
  if (i < n) p[i] = 0.f;
}

// ---------------- segment_sum via hardware f32 atomics ----------------
__global__ void k_scatter(const float* __restrict__ edge, const int* __restrict__ recv,
                          float* __restrict__ agg, int E) {
  int e = blockIdx.x * 8 + (threadIdx.x >> 5);
  if (e >= E) return;
  int lane = threadIdx.x & 31;
  int r = recv[e];
  const float* src = edge + (size_t)e * LAT;
  float* dst = agg + (size_t)r * LAT;
#pragma unroll
  for (int j = 0; j < 8; ++j) {
    int c = lane + j * 32;   // coalesced across lanes
    __hip_atomic_fetch_add(dst + c, src[c], __ATOMIC_RELAXED, __HIP_MEMORY_SCOPE_AGENT);
  }
}

// ---------------- staged loaders ----------------
DEV AStage g1_load_a(const float* A0, const float* A1, const float* A2,
                     int row, int i1, int i2, bool rok, int K0, int K1, int K2,
                     int Ktot, int kbase) {
  const float* src = nullptr;
  int rem = 0;
  if (rok) {   // a 32-run never straddles concat segments (multiples of 256, or K0=4)
    if (kbase < K0) {
      src = A0 + (size_t)row * K0 + kbase;            rem = K0 - kbase;
    } else if (kbase < K0 + K1) {
      src = A1 + (size_t)i1 * K1 + (kbase - K0);      rem = K0 + K1 - kbase;
    } else if (kbase < Ktot) {
      src = A2 + (size_t)i2 * K2 + (kbase - K0 - K1); rem = Ktot - kbase;
    }
  }
  AStage st;
  if (src && rem >= 32) {
#pragma unroll
    for (int j = 0; j < 4; ++j) {
      v8bf s;
#pragma unroll
      for (int i = 0; i < 8; ++i) s[i] = (bf16_t)src[j * 8 + i];  // 8x b128 loads
      st.v[j] = s;
    }
  } else {
#pragma unroll
    for (int j = 0; j < 4; ++j) {
      v8bf s;
#pragma unroll
      for (int i = 0; i < 8; ++i) {
        int e = j * 8 + i;
        s[i] = (src && e < rem) ? (bf16_t)src[e] : (bf16_t)0.f;
      }
      st.v[j] = s;
    }
  }
  return st;
}

DEV BStage g1_load_b(const bf16_t* Wt, int tid, int Ktot, int c) {
  const bf16_t* src = Wt + (size_t)tid * Ktot + c * 64;
  BStage st;
  if (c * 64 + 64 <= Ktot) {
#pragma unroll
    for (int j = 0; j < 8; ++j) st.v[j] = *(const v8bf*)(src + j * 8);
  } else {
    int kbase = c * 64;
#pragma unroll
    for (int j = 0; j < 8; ++j) {
      v8bf s;
#pragma unroll
      for (int i = 0; i < 8; ++i) {
        int e = j * 8 + i;
        s[i] = (kbase + e < Ktot) ? src[e] : (bf16_t)0.f;
      }
      st.v[j] = s;
    }
  }
  return st;
}

// ---------------- GEMM-1: hidden = swish(concat(A0,A1[idx1],A2[idx2]) @ W1 + b1) ----------------
// Block: 256 thr = 8 waves. Tile 128(M) x 256(N); each wave: 16 rows x 16 n-tiles.
// K staged 64 at a time with LDS ping-pong: global loads of stage c+1 overlap
// the WMMA work of stage c; one barrier per stage.
__global__ __launch_bounds__(256) void k_gemm1(
    const float* __restrict__ A0, const float* __restrict__ A1,
    const float* __restrict__ A2, const int* __restrict__ idx1,
    const int* __restrict__ idx2, int K0, int K1, int K2, int R,
    const bf16_t* __restrict__ Wt, const float* __restrict__ bias,
    bf16_t* __restrict__ out) {
  __shared__ bf16_t Alds[2][128 * 64];   // 2 x 16 KB
  __shared__ bf16_t Blds[2][256 * 64];   // 2 x 32 KB
  const int tid = threadIdx.x, lane = tid & 31, wv = tid >> 5;
  const int row0 = blockIdx.x * 128;
  const int Ktot = K0 + K1 + K2;
  const int NC = (Ktot + 63) >> 6;

  v8f acc[16];
#pragma unroll
  for (int t = 0; t < 16; ++t)
#pragma unroll
    for (int i = 0; i < 8; ++i) acc[t][i] = 0.f;

  // A loader mapping: 2 threads/row, 32 elems each (fp32 -> bf16 convert).
  const int lr = tid >> 1, lkk = (tid & 1) * 32;
  const int lrow = row0 + lr;
  const bool rok = lrow < R;
  const int i1 = (idx1 && rok) ? idx1[lrow] : lrow;
  const int i2 = (idx2 && rok) ? idx2[lrow] : lrow;

  AStage as = g1_load_a(A0, A1, A2, lrow, i1, i2, rok, K0, K1, K2, Ktot, lkk);
  BStage bs = g1_load_b(Wt, tid, Ktot, 0);
  stage_store4(Alds[0] + lr * 64 + lkk, as);
  stage_store8(Blds[0] + tid * 64, bs);
  __syncthreads();

  for (int c = 0; c < NC; ++c) {
    const int cur = c & 1;
    const bool more = (c + 1 < NC);
    if (more) {   // issue next-stage global loads before the matrix work
      as = g1_load_a(A0, A1, A2, lrow, i1, i2, rok, K0, K1, K2, Ktot,
                     (c + 1) * 64 + lkk);
      bs = g1_load_b(Wt, tid, Ktot, c + 1);
    }
    const bf16_t* Ab = Alds[cur];
    const bf16_t* Bb = Blds[cur];
#pragma unroll
    for (int kk = 0; kk < 64; kk += 32) {
      v16bf a = frag_a64(Ab, wv * 16, kk, lane);
      mma_row<16>(a, Bb, kk, lane, acc);
    }
    if (more) {
      stage_store4(Alds[cur ^ 1] + lr * 64 + lkk, as);
      stage_store8(Blds[cur ^ 1] + tid * 64, bs);
      __syncthreads();
    }
  }

  // epilogue: bias + swish -> bf16
  const int half = lane >> 4, lc = lane & 15;
#pragma unroll
  for (int r = 0; r < 8; ++r) {
    int row = row0 + wv * 16 + r + half * 8;
    if (row < R) {
#pragma unroll
      for (int t = 0; t < 16; ++t) {
        int n = t * 16 + lc;
        float x = acc[t][r] + bias[n];
        float sw = x / (1.f + __expf(-x));
        out[(size_t)row * 256 + n] = (bf16_t)sw;
      }
    }
  }
}

// ---------------- GEMM-2: out = [resid +] {LN}(hidden @ W2 + b2) ----------------
// MODE: 0 = bias only (decoder), 1 = bias+LN, 2 = bias+LN+residual (in place OK)
template <int NT, int MODE>
__global__ __launch_bounds__(256) void k_gemm2(
    const bf16_t* __restrict__ A, int R, const bf16_t* __restrict__ Wt,
    const float* __restrict__ bias, const float* __restrict__ lns,
    const float* __restrict__ lno, const float* __restrict__ resid,
    float* __restrict__ out) {
  constexpr int NOUT = NT * 16;
  __shared__ bf16_t Alds[2][128 * 64];
  __shared__ bf16_t Blds[2][NOUT * 64];
  const int tid = threadIdx.x, lane = tid & 31, wv = tid >> 5;
  const int row0 = blockIdx.x * 128;

  v8f acc[NT];
#pragma unroll
  for (int t = 0; t < NT; ++t)
#pragma unroll
    for (int i = 0; i < 8; ++i) acc[t][i] = 0.f;

  const int lr = tid >> 1, lkk = (tid & 1) * 32;
  const int lrow = row0 + lr;
  const bool rok = lrow < R;

  auto load_a = [&](int c) -> AStage {
    AStage st;
    if (rok) {
      const bf16_t* src = A + (size_t)lrow * 256 + c * 64 + lkk;
#pragma unroll
      for (int j = 0; j < 4; ++j) st.v[j] = *(const v8bf*)(src + j * 8);
    } else {
      v8bf z;
#pragma unroll
      for (int i = 0; i < 8; ++i) z[i] = (bf16_t)0.f;
#pragma unroll
      for (int j = 0; j < 4; ++j) st.v[j] = z;
    }
    return st;
  };
  auto load_b = [&](int c) -> BStage {
    BStage st;
    const bf16_t* src = Wt + (size_t)(tid < NOUT ? tid : 0) * 256 + c * 64;
#pragma unroll
    for (int j = 0; j < 8; ++j) st.v[j] = *(const v8bf*)(src + j * 8);
    return st;
  };

  AStage as = load_a(0);
  BStage bs = load_b(0);
  stage_store4(Alds[0] + lr * 64 + lkk, as);
  if (tid < NOUT) stage_store8(Blds[0] + tid * 64, bs);
  __syncthreads();

  for (int c = 0; c < 4; ++c) {  // K = 256 fixed, staged 64 at a time
    const int cur = c & 1;
    const bool more = (c + 1 < 4);
    if (more) {
      as = load_a(c + 1);
      bs = load_b(c + 1);
    }
    const bf16_t* Ab = Alds[cur];
    const bf16_t* Bb = Blds[cur];
#pragma unroll
    for (int kk = 0; kk < 64; kk += 32) {
      v16bf a = frag_a64(Ab, wv * 16, kk, lane);
      mma_row<NT>(a, Bb, kk, lane, acc);
    }
    if (more) {
      stage_store4(Alds[cur ^ 1] + lr * 64 + lkk, as);
      if (tid < NOUT) stage_store8(Blds[cur ^ 1] + tid * 64, bs);
      __syncthreads();
    }
  }

  const int half = lane >> 4, lc = lane & 15;
  float bs_[NT], ls[NT], lo_[NT];
#pragma unroll
  for (int t = 0; t < NT; ++t) {
    int n = t * 16 + lc;
    bs_[t] = bias[n];
    if (MODE >= 1) { ls[t] = lns[n]; lo_[t] = lno[n]; }
  }

  if (MODE == 0) {
#pragma unroll
    for (int r = 0; r < 8; ++r) {
      int row = row0 + wv * 16 + r + half * 8;
      if (row < R) {
#pragma unroll
        for (int t = 0; t < NT; ++t)
          out[(size_t)row * NOUT + t * 16 + lc] = acc[t][r] + bs_[t];
      }
    }
  } else {
#pragma unroll
    for (int r = 0; r < 8; ++r) {
      // each wave owns complete rows: LN is a 16-lane-half reduction
      float s = 0.f, q = 0.f;
#pragma unroll
      for (int t = 0; t < NT; ++t) {
        float x = acc[t][r] + bs_[t];
        s += x; q += x * x;
      }
#pragma unroll
      for (int off = 8; off >= 1; off >>= 1) {
        s += __shfl_xor(s, off, 32);
        q += __shfl_xor(q, off, 32);
      }
      float mean = s * (1.f / NOUT);
      float var = q * (1.f / NOUT) - mean * mean;
      float rstd = rsqrtf(var + 1e-5f);
      int row = row0 + wv * 16 + r + half * 8;
      if (row < R) {
#pragma unroll
        for (int t = 0; t < NT; ++t) {
          int n = t * 16 + lc;
          float x = acc[t][r] + bs_[t];
          float y = (x - mean) * rstd * ls[t] + lo_[t];
          if (MODE == 2) y += resid[(size_t)row * 256 + n];
          out[(size_t)row * 256 + n] = y;
        }
      }
    }
  }
}

// ---------------- host orchestration ----------------
extern "C" void kernel_launch(void* const* d_in, const int* in_sizes, int n_in,
                              void* d_out, int out_size, void* d_ws, size_t ws_size,
                              hipStream_t stream) {
  (void)in_sizes; (void)n_in; (void)out_size; (void)ws_size;
  const float* node_f = (const float*)d_in[0];
  const float* edge_f = (const float*)d_in[1];
  const int* senders = (const int*)d_in[2];
  const int* receivers = (const int*)d_in[3];
  const float* en_w1 = (const float*)d_in[4];  const float* en_b1 = (const float*)d_in[5];
  const float* en_w2 = (const float*)d_in[6];  const float* en_b2 = (const float*)d_in[7];
  const float* en_ls = (const float*)d_in[8];  const float* en_lo = (const float*)d_in[9];
  const float* ee_w1 = (const float*)d_in[10]; const float* ee_b1 = (const float*)d_in[11];
  const float* ee_w2 = (const float*)d_in[12]; const float* ee_b2 = (const float*)d_in[13];
  const float* ee_ls = (const float*)d_in[14]; const float* ee_lo = (const float*)d_in[15];
  const float* pe_w1 = (const float*)d_in[16]; const float* pe_b1 = (const float*)d_in[17];
  const float* pe_w2 = (const float*)d_in[18]; const float* pe_b2 = (const float*)d_in[19];
  const float* pe_ls = (const float*)d_in[20]; const float* pe_lo = (const float*)d_in[21];
  const float* pn_w1 = (const float*)d_in[22]; const float* pn_b1 = (const float*)d_in[23];
  const float* pn_w2 = (const float*)d_in[24]; const float* pn_b2 = (const float*)d_in[25];
  const float* pn_ls = (const float*)d_in[26]; const float* pn_lo = (const float*)d_in[27];
  const float* dc_w1 = (const float*)d_in[28]; const float* dc_b1 = (const float*)d_in[29];
  const float* dc_w2 = (const float*)d_in[30]; const float* dc_b2 = (const float*)d_in[31];

  char* ws = (char*)d_ws;
  size_t off = 0;
  auto carve = [&](size_t bytes) -> void* {
    void* p = ws + off;
    off = (off + bytes + 255) & ~(size_t)255;
    return p;
  };
  float* node = (float*)carve((size_t)NNODE * LAT * 4);
  float* edge = (float*)carve((size_t)NEDGE * LAT * 4);
  float* agg  = (float*)carve((size_t)NNODE * LAT * 4);
  bf16_t* hidden = (bf16_t*)carve((size_t)NEDGE * 256 * 2);

  auto conv = [&](const float* W, int K, int N) -> bf16_t* {
    bf16_t* dst = (bf16_t*)carve((size_t)K * N * 2);
    int total = K * N;
    hipLaunchKernelGGL(k_w2bf16t, dim3((total + 255) / 256), dim3(256), 0, stream,
                       W, dst, K, N);
    return dst;
  };
  bf16_t* wt_en1 = conv(en_w1, 128, 256);
  bf16_t* wt_en2 = conv(en_w2, 256, 256);
  bf16_t* wt_ee1 = conv(ee_w1, 4, 256);
  bf16_t* wt_ee2 = conv(ee_w2, 256, 256);
  bf16_t* wt_pe1[STEPS], *wt_pe2[STEPS], *wt_pn1[STEPS], *wt_pn2[STEPS];
  for (int s = 0; s < STEPS; ++s) {
    wt_pe1[s] = conv(pe_w1 + (size_t)s * 768 * 256, 768, 256);
    wt_pe2[s] = conv(pe_w2 + (size_t)s * 256 * 256, 256, 256);
    wt_pn1[s] = conv(pn_w1 + (size_t)s * 512 * 256, 512, 256);
    wt_pn2[s] = conv(pn_w2 + (size_t)s * 256 * 256, 256, 256);
  }
  bf16_t* wt_d1 = conv(dc_w1, 256, 256);
  bf16_t* wt_d2 = conv(dc_w2, 256, 128);

  const dim3 blkN((NNODE + 127) / 128), blkE((NEDGE + 127) / 128), thr(256);
  auto g1 = [&](const float* A0, const float* A1, const float* A2,
                const int* x1, const int* x2, int K0, int K1, int K2, int R,
                const bf16_t* Wt, const float* b) {
    hipLaunchKernelGGL(k_gemm1, dim3((R + 127) / 128), thr, 0, stream,
                       A0, A1, A2, x1, x2, K0, K1, K2, R, Wt, b, hidden);
  };

  // ---- encoders ----
  g1(node_f, nullptr, nullptr, nullptr, nullptr, 128, 0, 0, NNODE, wt_en1, en_b1);
  hipLaunchKernelGGL((k_gemm2<16, 1>), blkN, thr, 0, stream,
                     hidden, NNODE, wt_en2, en_b2, en_ls, en_lo, (const float*)nullptr, node);
  g1(edge_f, nullptr, nullptr, nullptr, nullptr, 4, 0, 0, NEDGE, wt_ee1, ee_b1);
  hipLaunchKernelGGL((k_gemm2<16, 1>), blkE, thr, 0, stream,
                     hidden, NEDGE, wt_ee2, ee_b2, ee_ls, ee_lo, (const float*)nullptr, edge);

  // ---- processor steps ----
  for (int s = 0; s < STEPS; ++s) {
    // edge update: MLP(concat[edge, node[senders], node[receivers]]) + resid, LN
    g1(edge, node, node, senders, receivers, 256, 256, 256, NEDGE,
       wt_pe1[s], pe_b1 + s * 256);
    hipLaunchKernelGGL((k_gemm2<16, 2>), blkE, thr, 0, stream,
                       hidden, NEDGE, wt_pe2[s], pe_b2 + s * 256,
                       pe_ls + s * 256, pe_lo + s * 256, edge, edge);
    // agg = segment_sum(edge, receivers)
    hipLaunchKernelGGL(k_fill0, dim3((NNODE * LAT + 255) / 256), thr, 0, stream,
                       agg, NNODE * LAT);
    hipLaunchKernelGGL(k_scatter, dim3((NEDGE + 7) / 8), thr, 0, stream,
                       edge, receivers, agg, NEDGE);
    // node update: MLP(concat[node, agg]) + resid, LN
    g1(node, agg, nullptr, nullptr, nullptr, 256, 256, 0, NNODE,
       wt_pn1[s], pn_b1 + s * 256);
    hipLaunchKernelGGL((k_gemm2<16, 2>), blkN, thr, 0, stream,
                       hidden, NNODE, wt_pn2[s], pn_b2 + s * 256,
                       pn_ls + s * 256, pn_lo + s * 256, node, node);
  }

  // ---- decoder (no LN) ----
  g1(node, nullptr, nullptr, nullptr, nullptr, 256, 0, 0, NNODE, wt_d1, dc_b1);
  hipLaunchKernelGGL((k_gemm2<8, 0>), blkN, thr, 0, stream,
                     hidden, NNODE, wt_d2, dc_b2, (const float*)nullptr,
                     (const float*)nullptr, (const float*)nullptr, (float*)d_out);
}
